// GMNLayer_54657753809036
// MI455X (gfx1250) — matrix operand order repaired and checked by reading, not compiled, not run on hardware
//
#include <hip/hip_runtime.h>

#define HID    128
#define NTILES 8

typedef __attribute__((ext_vector_type(16))) __bf16 bf16x16;
typedef __attribute__((ext_vector_type(8)))  float  f32x8;
typedef __attribute__((ext_vector_type(4)))  int    i32x4;

union ABFrag { bf16x16 v; uint4 u[2]; };
union Pack8  { __bf16 b[8]; uint4 q; };

// ---------------- LDS layout (shared by edge & node kernels), bytes ------------
// [0, 131072)       : three swizzled bf16 weight tables (8kb + 4kb + 4kb blocks)
// [131072, 133632)  : fp32 bias / small-vector area (5 x 128 floats)
// [133632, 201216)  : per-wave A staging, 16 rows x 528 B (264 bf16, 8 pad)
// [201216, 236032)  : per-wave A2 staging, 16 rows x 272 B (136 bf16, 8 pad)
// [236032, 240128)  : per-wave meta, 512 B
#define OFF_BIAS   131072
#define OFF_ASTAGE 133632
#define OFF_A2     201216
#define OFF_META   236032
#define LDS_TOTAL  240128
#define AST_BYTES  8448
#define A2_BYTES   4352
#define ASTRIDE    264     // bf16 elems per A-stage row (padded: 4m bank offset)
#define A2STRIDE   136     // bf16 elems per A2 row
#define GSTRIDE    132     // f32 elems per G row (reuses the 8448-B A-stage slot)

// Same-wave LDS ordering fence (DS ops are in-order per wave; this pins the
// compiler's scheduling across the staging<->fragment-load phase boundaries).
#define WAVE_LDS_SYNC() asm volatile("s_wait_dscnt 0x0" ::: "memory")

#if defined(__has_builtin)
#if __has_builtin(__builtin_amdgcn_global_load_async_to_lds_b128)
#define HAVE_ASYNC_LDS 1
#endif
#endif

struct EdgeMeta {
  int   row[16];
  int   valid[16];
  float radial[16];
  float dif[16][3];
};

// Guaranteed-hardware fp32 scatter-add (agent scope, relaxed, no return value
// => GLOBAL_ATOMIC_ADD_F32 without return, tracked on STOREcnt).
__device__ __forceinline__ void atomic_add_f32(float* p, float v) {
  __hip_atomic_fetch_add(p, v, __ATOMIC_RELAXED, __HIP_MEMORY_SCOPE_AGENT);
}

// B fragment: pre-swizzled so each lane's 16 bf16 (= K chunks [8h,8h+7] and [16+8h,16+8h+7] for its N
// column, per the 16-bit B-matrix VGPR layout) are 32 contiguous bytes in LDS.
__device__ __forceinline__ bf16x16 load_b_frag(const __bf16* sw, int kb, int nt, int lane) {
  const __bf16* q = sw + (((kb * NTILES + nt) * 32 + lane) << 4);
  ABFrag f;
  f.u[0] = *(const uint4*)(q);
  f.u[1] = *(const uint4*)(q + 8);
  return f.v;
}

// A fragment from row-major bf16 rows (16-bit A 16x32 layout):
// lane L: M = L&15, K-chunks [s, s+7] and [s+16, s+23], s = kb*32 + (L>=16 ? 8 : 0).
__device__ __forceinline__ bf16x16 load_a_frag(const __bf16* rows, int strideElems,
                                               int kb, int lane) {
  const int m = lane & 15;
  const int s = kb * 32 + ((lane >> 4) << 3);
  const __bf16* p = rows + m * strideElems + s;
  ABFrag f;
  f.u[0] = *(const uint4*)(p);
  f.u[1] = *(const uint4*)(p + 16);
  return f.v;
}

__device__ __forceinline__ f32x8 wmma_bf16(bf16x16 a, bf16x16 b, f32x8 c) {
  return __builtin_amdgcn_wmma_f32_16x16x32_bf16(false, a, false, b, (short)0, c,
                                                 false, false);
}

// Stage the 131072-B swizzled weight block global->LDS.
// Preferred path: GLOBAL_LOAD_ASYNC_TO_LDS_B128 (ASYNCcnt-tracked DMA into LDS,
// no VGPR bounce); fallback: vectorized load+ds_store copy.
__device__ __forceinline__ void stage_weights(const unsigned short* __restrict__ wsw,
                                              char* smem, int tid) {
#if defined(HAVE_ASYNC_LDS)
  const char* g = (const char*)wsw;
#pragma unroll
  for (int i = 0; i < 32; ++i) {
    const int byteOff = (tid + 256 * i) * 16;
    void* gp = (void*)(g + byteOff);
    void* lp = (void*)(smem + byteOff);
    __builtin_amdgcn_global_load_async_to_lds_b128(
        (__attribute__((address_space(1))) i32x4*)gp,
        (__attribute__((address_space(3))) i32x4*)lp,
        0, 0);
  }
  asm volatile("s_wait_asynccnt 0x0" ::: "memory");  // own-wave asyncs done
#else
  const uint4* src = (const uint4*)wsw;
  uint4* dst = (uint4*)smem;
#pragma unroll
  for (int i = 0; i < 32; ++i) dst[tid + 256 * i] = src[tid + 256 * i];
#endif
}

__device__ __forceinline__ void swz_decode(int t, int& k, int& n) {
  const int jj   = t & 15;
  const int lane = (t >> 4) & 31;
  const int nt   = (t >> 9) & 7;
  const int kb   = t >> 12;
  k = kb * 32 + ((lane >> 4) << 3) + (jj & 7) + ((jj >> 3) << 4);   // K chunks [8h,8h+7] and [16+8h,16+8h+7]
  n = nt * 16 + (lane & 15);
}

// ---- Pre-convert + pre-swizzle all GEMM weights to bf16 B-fragment order ----
__global__ void egnn_weights_kernel(const float* __restrict__ we1,
                                    const float* __restrict__ we2,
                                    const float* __restrict__ wc1,
                                    const float* __restrict__ wn1,
                                    const float* __restrict__ wn2,
                                    const float* __restrict__ wv1,
                                    unsigned short* __restrict__ wedge,
                                    unsigned short* __restrict__ wnode) {
  const int i = blockIdx.x * 256 + threadIdx.x;
  if (i >= 131072) return;
  float val;
  unsigned short* dst;
  if (i < 65536) {                       // edge tables: we1(K=256) | we2 | wc1
    const int j = i;
    int k, n;
    if (j < 32768)      { swz_decode(j, k, n);         val = we1[k * HID + n]; }
    else if (j < 49152) { swz_decode(j - 32768, k, n); val = we2[k * HID + n]; }
    else                { swz_decode(j - 49152, k, n); val = wc1[k * HID + n]; }
    dst = wedge + j;
  } else {                               // node tables: wn1_eff(K=256) | wn2 | wv1
    const int j = i - 65536;
    int k, n;
    if (j < 32768) {
      swz_decode(j, k, n);
      // n_in = [h, h, agg]  =>  wn1_eff[k] = wn1[k] + wn1[k+128] (k<128), else wn1[k+128]
      val = (k < HID) ? (wn1[k * HID + n] + wn1[(k + HID) * HID + n])
                      : wn1[(k + HID) * HID + n];
    } else if (j < 49152) { swz_decode(j - 32768, k, n); val = wn2[k * HID + n]; }
    else                  { swz_decode(j - 49152, k, n); val = wv1[k * HID + n]; }
    dst = wnode + j;
  }
  union { __bf16 b; unsigned short s; } cv;
  cv.b = (__bf16)val;
  *dst = cv.s;
}

__global__ void egnn_zero_kernel(float* __restrict__ p, long long n) {
  long long i = (long long)blockIdx.x * blockDim.x + threadIdx.x;
  const long long stride = (long long)gridDim.x * blockDim.x;
  for (; i < n; i += stride) p[i] = 0.0f;
}

// ---------------- Edge kernel: 128 edges / block, 16 edges / wave ------------
__global__ __launch_bounds__(256, 1) void egnn_edge_kernel(
    const float* __restrict__ h, const int* __restrict__ ei,
    const float* __restrict__ x,
    const float* __restrict__ we1, const float* __restrict__ be1,
    const float* __restrict__ be2, const float* __restrict__ bc1,
    const float* __restrict__ wc2,
    const unsigned short* __restrict__ wsw,
    float* __restrict__ agg, float* __restrict__ numv, float* __restrict__ cnt,
    int nEdges) {
  extern __shared__ char smem[];
  const int tid  = threadIdx.x;
  const int lane = tid & 31;
  const int wave = tid >> 5;

  stage_weights(wsw, smem, tid);
  float* ldsf = (float*)(smem + OFF_BIAS);
  if (tid < HID) {
    ldsf[tid]           = be1[tid];
    ldsf[HID + tid]     = be2[tid];
    ldsf[2 * HID + tid] = bc1[tid];
    ldsf[3 * HID + tid] = we1[256 * HID + tid];  // radial row of we1 (K=256)
    ldsf[4 * HID + tid] = wc2[tid];
  }
  __syncthreads();

  const __bf16* WE1 = (const __bf16*)smem;
  const __bf16* WE2 = WE1 + 32768;
  const __bf16* WC1 = WE1 + 49152;
  const float* Lbe1 = ldsf;
  const float* Lbe2 = ldsf + HID;
  const float* Lbc1 = ldsf + 2 * HID;
  const float* Lw1r = ldsf + 3 * HID;
  const float* Lwc2 = ldsf + 4 * HID;

  __bf16* Aw  = (__bf16*)(smem + OFF_ASTAGE + wave * AST_BYTES);
  __bf16* A2w = (__bf16*)(smem + OFF_A2 + wave * A2_BYTES);
  float*  Gw  = (float*)(smem + OFF_ASTAGE + wave * AST_BYTES);  // reuse
  EdgeMeta* mw = (EdgeMeta*)(smem + OFF_META + wave * 512);

  const int  m  = lane & 15;
  const int  e  = blockIdx.x * 128 + wave * 16 + m;
  const bool ev = (e < nEdges);
  const int  eC = ev ? e : (nEdges - 1);

  // Gather: lanes 0-15 copy h[row] into cols [0,128), lanes 16-31 h[col] into [128,256)
  {
    const int node = (lane < 16) ? ei[eC] : ei[nEdges + eC];
    const float4* hs = (const float4*)(h + (size_t)node * HID);
    __bf16* dstRow = Aw + m * ASTRIDE + ((lane < 16) ? 0 : HID);
#pragma unroll
    for (int i = 0; i < 16; ++i) {
      float4 f0 = hs[2 * i + 0];
      float4 f1 = hs[2 * i + 1];
      Pack8 pk;
      pk.b[0] = (__bf16)f0.x; pk.b[1] = (__bf16)f0.y;
      pk.b[2] = (__bf16)f0.z; pk.b[3] = (__bf16)f0.w;
      pk.b[4] = (__bf16)f1.x; pk.b[5] = (__bf16)f1.y;
      pk.b[6] = (__bf16)f1.z; pk.b[7] = (__bf16)f1.w;
      *(uint4*)(dstRow + 8 * i) = pk.q;
    }
    if (lane < 16) {
      const int r = ei[eC];
      const int c = ei[nEdges + eC];
      const float dx = x[3 * r + 0] - x[3 * c + 0];
      const float dy = x[3 * r + 1] - x[3 * c + 1];
      const float dz = x[3 * r + 2] - x[3 * c + 2];
      mw->row[m] = r;
      mw->valid[m] = ev ? 1 : 0;
      mw->radial[m] = dx * dx + dy * dy + dz * dz;
      mw->dif[m][0] = dx; mw->dif[m][1] = dy; mw->dif[m][2] = dz;
    }
  }
  WAVE_LDS_SYNC();

  const int hi8 = (lane >> 4) << 3;
  const int nl  = lane & 15;

  // ---- edge MLP layer 1: relu([h_row|h_col|radial] @ we1 + be1) -> A2 (bf16) ----
  {
    f32x8 acc[NTILES] = {};
#pragma unroll
    for (int kb = 0; kb < 8; ++kb) {
      bf16x16 a = load_a_frag(Aw, ASTRIDE, kb, lane);
#pragma unroll
      for (int nt = 0; nt < NTILES; ++nt)
        acc[nt] = wmma_bf16(a, load_b_frag(WE1, kb, nt, lane), acc[nt]);
    }
#pragma unroll
    for (int r = 0; r < 8; ++r) {
      const int mm = r + hi8;
      const float rad = mw->radial[mm];
#pragma unroll
      for (int nt = 0; nt < NTILES; ++nt) {
        const int n = nt * 16 + nl;
        float vv = acc[nt][r] + rad * Lw1r[n] + Lbe1[n];  // rank-1 radial column
        vv = fmaxf(vv, 0.0f);
        A2w[mm * A2STRIDE + n] = (__bf16)vv;
      }
    }
  }
  WAVE_LDS_SYNC();

  // ---- layer 2 -> edge_feat; scatter-add agg; keep bf16 copy in A2 ----
  {
    f32x8 acc[NTILES] = {};
#pragma unroll
    for (int kb = 0; kb < 4; ++kb) {
      bf16x16 a = load_a_frag(A2w, A2STRIDE, kb, lane);
#pragma unroll
      for (int nt = 0; nt < NTILES; ++nt)
        acc[nt] = wmma_bf16(a, load_b_frag(WE2, kb, nt, lane), acc[nt]);
    }
#pragma unroll
    for (int r = 0; r < 8; ++r) {
      const int mm = r + hi8;
      const int rowm = mw->row[mm];
      const int vmask = mw->valid[mm];
#pragma unroll
      for (int nt = 0; nt < NTILES; ++nt) {
        const int n = nt * 16 + nl;
        const float vv = fmaxf(acc[nt][r] + Lbe2[n], 0.0f);
        if (vmask) atomic_add_f32(agg + (size_t)rowm * HID + n, vv);
        A2w[mm * A2STRIDE + n] = (__bf16)vv;
      }
    }
  }
  WAVE_LDS_SYNC();

  // ---- coord gate: relu(EF @ wc1 + bc1) -> Gscr (f32, reuses A-stage slot) ----
  {
    f32x8 acc[NTILES] = {};
#pragma unroll
    for (int kb = 0; kb < 4; ++kb) {
      bf16x16 a = load_a_frag(A2w, A2STRIDE, kb, lane);
#pragma unroll
      for (int nt = 0; nt < NTILES; ++nt)
        acc[nt] = wmma_bf16(a, load_b_frag(WC1, kb, nt, lane), acc[nt]);
    }
#pragma unroll
    for (int r = 0; r < 8; ++r) {
      const int mm = r + hi8;
#pragma unroll
      for (int nt = 0; nt < NTILES; ++nt) {
        const int n = nt * 16 + nl;
        Gw[mm * GSTRIDE + n] = fmaxf(acc[nt][r] + Lbc1[n], 0.0f);
      }
    }
  }
  WAVE_LDS_SYNC();

  // gate = G @ wc2 ; trans = clip(diff * gate) ; scatter num / cnt
  if (lane < 16 && mw->valid[m]) {
    float g = 0.0f;
    const float* grow = Gw + m * GSTRIDE;
#pragma unroll 8
    for (int n = 0; n < HID; ++n) g += grow[n] * Lwc2[n];
    const int rowm = mw->row[m];
#pragma unroll
    for (int d = 0; d < 3; ++d) {
      float t = mw->dif[m][d] * g;
      t = fminf(fmaxf(t, -100.0f), 100.0f);
      atomic_add_f32(numv + (size_t)rowm * 3 + d, t);
    }
    atomic_add_f32(cnt + rowm, 1.0f);
  }
}

// ---------------- Node kernel: 128 nodes / block, 16 nodes / wave ------------
__global__ __launch_bounds__(256, 1) void egnn_node_kernel(
    const float* __restrict__ h, const float* __restrict__ x,
    const float* __restrict__ v,
    const float* __restrict__ bn1, const float* __restrict__ bn2,
    const float* __restrict__ bv1, const float* __restrict__ wv2,
    const float* __restrict__ bv2,
    const unsigned short* __restrict__ wsw,
    const float* __restrict__ agg, const float* __restrict__ numv,
    const float* __restrict__ cnt,
    float* __restrict__ out_h, float* __restrict__ out_x,
    float* __restrict__ out_v, int nNodes) {
  extern __shared__ char smem[];
  const int tid  = threadIdx.x;
  const int lane = tid & 31;
  const int wave = tid >> 5;

  stage_weights(wsw, smem, tid);
  float* ldsf = (float*)(smem + OFF_BIAS);
  if (tid < HID) {
    ldsf[tid]           = bn1[tid];
    ldsf[HID + tid]     = bn2[tid];
    ldsf[2 * HID + tid] = bv1[tid];
    ldsf[3 * HID + tid] = wv2[tid];
  }
  if (tid == 0) ldsf[4 * HID] = bv2[0];
  __syncthreads();

  const __bf16* WN1 = (const __bf16*)smem;   // wn1_eff, K=256
  const __bf16* WN2 = WN1 + 32768;
  const __bf16* WV1 = WN1 + 49152;
  const float* Lbn1 = ldsf;
  const float* Lbn2 = ldsf + HID;
  const float* Lbv1 = ldsf + 2 * HID;
  const float* Lwv2 = ldsf + 3 * HID;
  const float  vbias = ldsf[4 * HID];

  __bf16* Aw  = (__bf16*)(smem + OFF_ASTAGE + wave * AST_BYTES);
  __bf16* A2w = (__bf16*)(smem + OFF_A2 + wave * A2_BYTES);
  float*  Gw  = (float*)(smem + OFF_ASTAGE + wave * AST_BYTES);  // reuse

  const int  m     = lane & 15;
  const int  node0 = blockIdx.x * 128 + wave * 16;
  const int  nodeM = node0 + m;
  const bool nv    = (nodeM < nNodes);
  const int  nodeC = nv ? nodeM : (nNodes - 1);

  // Gather: lanes 0-15 -> h[node] into [0,128); lanes 16-31 -> agg[node] into [128,256)
  {
    const float4* src = (lane < 16) ? (const float4*)(h + (size_t)nodeC * HID)
                                    : (const float4*)(agg + (size_t)nodeC * HID);
    __bf16* dstRow = Aw + m * ASTRIDE + ((lane < 16) ? 0 : HID);
#pragma unroll
    for (int i = 0; i < 16; ++i) {
      float4 f0 = src[2 * i + 0];
      float4 f1 = src[2 * i + 1];
      Pack8 pk;
      pk.b[0] = (__bf16)f0.x; pk.b[1] = (__bf16)f0.y;
      pk.b[2] = (__bf16)f0.z; pk.b[3] = (__bf16)f0.w;
      pk.b[4] = (__bf16)f1.x; pk.b[5] = (__bf16)f1.y;
      pk.b[6] = (__bf16)f1.z; pk.b[7] = (__bf16)f1.w;
      *(uint4*)(dstRow + 8 * i) = pk.q;
    }
  }
  WAVE_LDS_SYNC();

  const int hi8 = (lane >> 4) << 3;
  const int nl  = lane & 15;

  // node MLP layer 1: relu([h|agg] @ wn1_eff + bn1) -> A2 (bf16)
  {
    f32x8 acc[NTILES] = {};
#pragma unroll
    for (int kb = 0; kb < 8; ++kb) {
      bf16x16 a = load_a_frag(Aw, ASTRIDE, kb, lane);
#pragma unroll
      for (int nt = 0; nt < NTILES; ++nt)
        acc[nt] = wmma_bf16(a, load_b_frag(WN1, kb, nt, lane), acc[nt]);
    }
#pragma unroll
    for (int r = 0; r < 8; ++r) {
      const int mm = r + hi8;
#pragma unroll
      for (int nt = 0; nt < NTILES; ++nt) {
        const int n = nt * 16 + nl;
        A2w[mm * A2STRIDE + n] = (__bf16)fmaxf(acc[nt][r] + Lbn1[n], 0.0f);
      }
    }
  }
  WAVE_LDS_SYNC();

  // node MLP layer 2 + residual: h_new = h + (A2 @ wn2 + bn2)
  {
    f32x8 acc[NTILES] = {};
#pragma unroll
    for (int kb = 0; kb < 4; ++kb) {
      bf16x16 a = load_a_frag(A2w, A2STRIDE, kb, lane);
#pragma unroll
      for (int nt = 0; nt < NTILES; ++nt)
        acc[nt] = wmma_bf16(a, load_b_frag(WN2, kb, nt, lane), acc[nt]);
    }
#pragma unroll
    for (int r = 0; r < 8; ++r) {
      const int mm = r + hi8;
      const int nd = node0 + mm;
      if (nd < nNodes) {
#pragma unroll
        for (int nt = 0; nt < NTILES; ++nt) {
          const int n = nt * 16 + nl;
          out_h[(size_t)nd * HID + n] =
              h[(size_t)nd * HID + n] + acc[nt][r] + Lbn2[n];
        }
      }
    }
  }

  // velocity gate layer 1: relu(h @ wv1 + bv1) -> Gscr (overwrites A-stage slot)
  {
    f32x8 acc[NTILES] = {};
#pragma unroll
    for (int kb = 0; kb < 4; ++kb) {
      bf16x16 a = load_a_frag(Aw, ASTRIDE, kb, lane);  // first 128 cols = h
#pragma unroll
      for (int nt = 0; nt < NTILES; ++nt)
        acc[nt] = wmma_bf16(a, load_b_frag(WV1, kb, nt, lane), acc[nt]);
    }
#pragma unroll
    for (int r = 0; r < 8; ++r) {
      const int mm = r + hi8;
#pragma unroll
      for (int nt = 0; nt < NTILES; ++nt) {
        const int n = nt * 16 + nl;
        Gw[mm * GSTRIDE + n] = fmaxf(acc[nt][r] + Lbv1[n], 0.0f);
      }
    }
  }
  WAVE_LDS_SYNC();

  if (lane < 16 && nv) {
    float vg = vbias;
    const float* grow = Gw + m * GSTRIDE;
#pragma unroll 8
    for (int n = 0; n < HID; ++n) vg += grow[n] * Lwv2[n];
    const float c = fmaxf(cnt[nodeM], 1.0f);
#pragma unroll
    for (int d = 0; d < 3; ++d) {
      const float f  = numv[(size_t)nodeM * 3 + d] / c;
      const float vn = vg * v[(size_t)nodeM * 3 + d] + f;
      out_v[(size_t)nodeM * 3 + d] = vn;
      out_x[(size_t)nodeM * 3 + d] = x[(size_t)nodeM * 3 + d] + vn;
    }
  }
}

extern "C" void kernel_launch(void* const* d_in, const int* in_sizes, int n_in,
                              void* d_out, int out_size, void* d_ws, size_t ws_size,
                              hipStream_t stream) {
  (void)n_in; (void)out_size; (void)ws_size;
  const float* h   = (const float*)d_in[0];
  const int*   ei  = (const int*)  d_in[1];
  const float* x   = (const float*)d_in[2];
  const float* v   = (const float*)d_in[3];
  const float* we1 = (const float*)d_in[4];
  const float* be1 = (const float*)d_in[5];
  const float* we2 = (const float*)d_in[6];
  const float* be2 = (const float*)d_in[7];
  const float* wn1 = (const float*)d_in[8];
  const float* bn1 = (const float*)d_in[9];
  const float* wn2 = (const float*)d_in[10];
  const float* bn2 = (const float*)d_in[11];
  const float* wv1 = (const float*)d_in[12];
  const float* bv1 = (const float*)d_in[13];
  const float* wv2 = (const float*)d_in[14];
  const float* bv2 = (const float*)d_in[15];
  const float* wc1 = (const float*)d_in[16];
  const float* bc1 = (const float*)d_in[17];
  const float* wc2 = (const float*)d_in[18];

  const int nNodes = in_sizes[0] / HID;   // 100000
  const int nEdges = in_sizes[1] / 2;     // 1600000

  char* ws = (char*)d_ws;
  unsigned short* wedge = (unsigned short*)ws;               // 131072 B
  unsigned short* wnode = (unsigned short*)(ws + 131072);    // 131072 B
  float* agg  = (float*)(ws + 262144);                       // [N,128] f32
  float* numv = agg + (size_t)nNodes * HID;                  // [N,3]   f32
  float* cnt  = numv + (size_t)nNodes * 3;                   // [N]     f32

  float* out_h = (float*)d_out;
  float* out_x = out_h + (size_t)nNodes * HID;
  float* out_v = out_x + (size_t)nNodes * 3;

  egnn_weights_kernel<<<512, 256, 0, stream>>>(we1, we2, wc1, wn1, wn2, wv1,
                                               wedge, wnode);
  egnn_zero_kernel<<<2048, 256, 0, stream>>>(agg, (long long)nNodes * (HID + 4));
  egnn_edge_kernel<<<(nEdges + 127) / 128, 256, LDS_TOTAL, stream>>>(
      h, ei, x, we1, be1, be2, bc1, wc2, wedge, agg, numv, cnt, nEdges);
  egnn_node_kernel<<<(nNodes + 127) / 128, 256, LDS_TOTAL, stream>>>(
      h, x, v, bn1, bn2, bv1, wv2, bv2, wnode, agg, numv, cnt,
      out_h, out_x, out_v, nNodes);
}
